// GraphEncoder_60533269070353
// MI455X (gfx1250) — matrix operand order repaired and checked by reading, not compiled, not run on hardware
//
#include <hip/hip_runtime.h>

// ---------------- constants ----------------
#define Dm   128
#define Nn   64
#define BTn  512
#define Rr   (BTn*Nn)      // 32768 rows
#define NH   4
#define DHh  32
#define NL   3
#define HID  512
#define LDK  136           // bf16 LDS row stride (16B-aligned rows, staggers banks)

typedef __bf16 bf16;
typedef __bf16 v16bf __attribute__((ext_vector_type(16)));
typedef float  v8f   __attribute__((ext_vector_type(8)));

static __device__ inline v8f v8zero() {
  v8f z;
#pragma unroll
  for (int j = 0; j < 8; j++) z[j] = 0.f;
  return z;
}

// ---- CDNA5 async global->LDS copy (ASYNCcnt tracked) ----
static __device__ __forceinline__ void async_b128(unsigned lds_off, const void* gaddr) {
  asm volatile("global_load_async_to_lds_b128 %0, %1, off"
               :: "v"(lds_off), "v"(gaddr)
               : "memory");
}
static __device__ __forceinline__ void wait_async0() {
  asm volatile("s_wait_asynccnt 0" ::: "memory");
}
// stage a 128x128 bf16 tile (pre-transposed in global, row stride = gstride elems)
// into LDS at ldsbase with padded row stride LDK
static __device__ __forceinline__ void stage_wt_async(const bf16* __restrict__ g, size_t gstride,
                                                      unsigned ldsbase, int tid) {
#pragma unroll
  for (int it = 0; it < 8; it++) {
    int c = tid + it * 256;            // 2048 x 16B chunks
    int n = c >> 4, j = c & 15;
    async_b128(ldsbase + (unsigned)(n * LDK + j * 8) * 2, g + (size_t)n * gstride + j * 8);
  }
}
// flat contiguous async copy (chunks of 16B)
static __device__ __forceinline__ void copy_async_flat(const void* g, unsigned ldsbase,
                                                       int chunks, int tid, int nt) {
  for (int c = tid; c < chunks; c += nt)
    async_b128(ldsbase + (unsigned)c * 16, (const char*)g + (size_t)c * 16);
}

// A fragment: 16x32 bf16 (M x K). lane: m=lane&15, koff=16*(lane>>4).
static __device__ inline v16bf load_afrag(const bf16* tile /* [16][LDK] */, int m, int lh, int k0) {
  v16bf af;
  const bf16* ar = tile + m * LDK + k0 + lh * 16;
#pragma unroll
  for (int j = 0; j < 16; j++) af[j] = ar[j];
  return af;
}
// B fragment: 32x16 bf16 (K x N) from W^T stored as WT[n][k].
static __device__ inline v16bf load_bfrag(const bf16* WT, int n, int lh, int k0) {
  v16bf bfr;
  const bf16* br = WT + n * LDK + k0 + lh * 16;
#pragma unroll
  for (int j = 0; j < 16; j++) bfr[j] = br[j];
  return bfr;
}

// ---------------- fp32 -> bf16 transposed weight convert: d[n*K+k] = (bf16)s[k*Ncol+n] ----------------
__global__ void k_f2bfT(const float* __restrict__ s, bf16* __restrict__ d, int K, int Ncol) {
  int i = blockIdx.x * 256 + threadIdx.x;
  if (i < K * Ncol) {
    int n = i / K, k = i - n * K;
    d[i] = (bf16)s[(size_t)k * Ncol + n];
  }
}

// ---------------- adjacency blend + edge bias + CSR rowptr ----------------
__global__ void k_adj(const float* __restrict__ P, const float* __restrict__ Q,
                      const float* __restrict__ alpha_p, const float* __restrict__ A0,
                      const int* __restrict__ src, const int* __restrict__ dst, int E,
                      float* __restrict__ A, float* __restrict__ logA0, int* __restrict__ rowptr) {
  int t = threadIdx.x;
  float alpha = alpha_p[0];
  if (t < Nn) {
    float row[Nn];
    float s = 0.f;
    for (int j = 0; j < Nn; j++) {
      float pq = 0.f;
#pragma unroll
      for (int r = 0; r < 8; r++) pq += P[t * 8 + r] * Q[j * 8 + r];
      float sp = (pq > 20.f) ? pq : log1pf(__expf(pq));      // softplus
      float a0 = A0[t * Nn + j];
      float m = (a0 > 0.f) ? 1.f : 0.f;
      float a = a0 * (1.f + alpha * sp * m);
      row[j] = a; s += a;
    }
    float inv = 1.f / (s + 1e-8f);
    for (int j = 0; j < Nn; j++) A[t * Nn + j] = row[j] * inv;
  }
  for (int e = t; e < E; e += blockDim.x) {
    logA0[e] = __logf(A0[src[e] * Nn + dst[e]] + 1e-8f);
    if (e == 0 || src[e] != src[e - 1]) rowptr[src[e]] = e;
  }
  if (t == 0) rowptr[Nn] = E;
}

// ---------------- LayerNorm: one wave32 per 128-wide row ----------------
__global__ void k_ln(const float* __restrict__ X, const float* __restrict__ g,
                     const float* __restrict__ b, float* __restrict__ Y) {
  int wave = threadIdx.x >> 5, lane = threadIdx.x & 31;
  size_t row = (size_t)blockIdx.x * 8 + wave;
  float4 v = ((const float4*)(X + row * Dm))[lane];
  float s = v.x + v.y + v.z + v.w;
#pragma unroll
  for (int o = 16; o; o >>= 1) s += __shfl_xor(s, o, 32);
  float mu = s * (1.f / Dm);
  float dx = v.x - mu, dy = v.y - mu, dz = v.z - mu, dw = v.w - mu;
  float q = dx * dx + dy * dy + dz * dz + dw * dw;
#pragma unroll
  for (int o = 16; o; o >>= 1) q += __shfl_xor(q, o, 32);
  float rs = rsqrtf(q * (1.f / Dm) + 1e-5f);
  float4 gg = ((const float4*)g)[lane];
  float4 bb = ((const float4*)b)[lane];
  float4 o4 = { dx * rs * gg.x + bb.x, dy * rs * gg.y + bb.y,
                dz * rs * gg.z + bb.z, dw * rs * gg.w + bb.w };
  ((float4*)(Y + row * Dm))[lane] = o4;
}

// ---------------- Hmix = A(64x64) @ Hn(64x128) per bt ----------------
__global__ void k_hmix(const float* __restrict__ Hn, const float* __restrict__ A,
                       float* __restrict__ Hm) {
  __shared__ float Hs[Nn][Dm];
  __shared__ float As[Nn][Nn];
  int bt = blockIdx.x;
  const float* Hb = Hn + (size_t)bt * Nn * Dm;
  copy_async_flat(Hb, (unsigned)(size_t)&Hs[0][0], Nn * Dm / 4, threadIdx.x, 256);
  copy_async_flat(A,  (unsigned)(size_t)&As[0][0], Nn * Nn / 4, threadIdx.x, 256);
  wait_async0();
  __syncthreads();
  float* Ho = Hm + (size_t)bt * Nn * Dm;
  for (int idx = threadIdx.x; idx < Nn * Dm; idx += 256) {
    int i = idx >> 7, d = idx & 127;
    float s = 0.f;
#pragma unroll
    for (int j = 0; j < Nn; j++) s += As[i][j] * Hs[j][d];
    Ho[idx] = s;
  }
}

// ---------------- generic 128-col WMMA GEMM: C = A(fp32)@W (W^T bf16 pre-transposed) ----------------
// grid.x = M/128, block = 256 (8 waves). K = 128, Ncol = 128.
__global__ void k_gemm(const float* __restrict__ Ain, const bf16* __restrict__ WtG,
                       const float* __restrict__ bias, const float* __restrict__ res1,
                       const float* __restrict__ res2, float* __restrict__ Cout, int flags) {
  extern __shared__ __align__(16) char smem[];
  bf16* Abf = (bf16*)smem;                     // [128][LDK]
  bf16* WT  = Abf + 128 * LDK;                 // [128][LDK]  (WT[n][k])
  int tid = threadIdx.x;
  size_t rowBase = (size_t)blockIdx.x * 128;

  // async DMA: W^T tile global -> LDS
  stage_wt_async(WtG, 128, (unsigned)(size_t)WT, tid);

  // A tile: fp32 -> bf16 convert through VGPRs
  const float4* Ag = (const float4*)(Ain + rowBase * Dm);
#pragma unroll
  for (int it = 0; it < 16; it++) {
    int f = tid + it * 256;                    // float4 idx, 4096 total
    int r = f >> 5, c4 = f & 31;
    float4 v = Ag[f];
    bf16* p = Abf + r * LDK + c4 * 4;
    p[0] = (bf16)v.x; p[1] = (bf16)v.y; p[2] = (bf16)v.z; p[3] = (bf16)v.w;
  }
  wait_async0();
  __syncthreads();

  int wave = tid >> 5, lane = tid & 31;
  int m = lane & 15, lh = lane >> 4;
  v8f acc[8];
#pragma unroll
  for (int ct = 0; ct < 8; ct++) acc[ct] = v8zero();

#pragma unroll
  for (int ks = 0; ks < 4; ks++) {
    int k0 = ks * 32;
    v16bf af = load_afrag(Abf + wave * 16 * LDK, m, lh, k0);
#pragma unroll
    for (int ct = 0; ct < 8; ct++) {
      v16bf bfr = load_bfrag(WT, ct * 16 + m, lh, k0);
      acc[ct] = __builtin_amdgcn_wmma_f32_16x16x32_bf16(false, af, false, bfr,
                                                        (short)0, acc[ct], false, false);
    }
  }
#pragma unroll
  for (int ct = 0; ct < 8; ct++) {
#pragma unroll
    for (int v = 0; v < 8; v++) {
      size_t rg = rowBase + wave * 16 + v + 8 * lh;
      int cg = ct * 16 + m;
      float val = acc[ct][v];
      if (flags & 1) val += bias[cg];
      size_t off = rg * Dm + cg;
      if (flags & 2) val += res1[off];
      if (flags & 4) val += res2[off];
      Cout[off] = val;
    }
  }
}

// ---------------- edge attention (GATv2 segment softmax), one block per bt ----------------
__global__ void k_attn(const float* __restrict__ Xq, const float* __restrict__ Xv,
                       const float* __restrict__ a_att, const float* __restrict__ logA0,
                       const int* __restrict__ rowptr, const int* __restrict__ dst,
                       float* __restrict__ Y) {
  extern __shared__ __align__(16) char smem[];
  float* Qs = (float*)smem;              // [64][128]
  float* Vs = Qs + Nn * Dm;              // [64][128]
  __shared__ int rp[Nn + 1];
  int bt = blockIdx.x, tid = threadIdx.x;
  const float* Qg = Xq + (size_t)bt * Nn * Dm;
  const float* Vg = Xv + (size_t)bt * Nn * Dm;
  copy_async_flat(Qg, (unsigned)(size_t)Qs, Nn * Dm / 4, tid, 256);
  copy_async_flat(Vg, (unsigned)(size_t)Vs, Nn * Dm / 4, tid, 256);
  if (tid <= Nn) rp[tid] = rowptr[tid];
  wait_async0();
  __syncthreads();

  int i = tid >> 2, h = tid & 3;
  float qi[DHh], av[DHh];
#pragma unroll
  for (int k = 0; k < DHh; k++) { qi[k] = Qs[i * Dm + h * DHh + k]; av[k] = a_att[h * DHh + k]; }
  int e0 = rp[i], e1 = rp[i + 1];

  float mx = -1e30f;
  for (int e = e0; e < e1; e++) {
    int d = dst[e];
    float s = 0.f;
#pragma unroll
    for (int k = 0; k < DHh; k++) {
      float x = qi[k] + Qs[d * Dm + h * DHh + k];
      x = (x > 0.f) ? x : 0.2f * x;
      s += x * av[k];
    }
    mx = fmaxf(mx, s + logA0[e]);
  }
  float den = 0.f, y[DHh];
#pragma unroll
  for (int k = 0; k < DHh; k++) y[k] = 0.f;
  for (int e = e0; e < e1; e++) {
    int d = dst[e];
    float s = 0.f;
#pragma unroll
    for (int k = 0; k < DHh; k++) {
      float x = qi[k] + Qs[d * Dm + h * DHh + k];
      x = (x > 0.f) ? x : 0.2f * x;
      s += x * av[k];
    }
    float w = __expf(s + logA0[e] - mx);
    den += w;
#pragma unroll
    for (int k = 0; k < DHh; k++) y[k] += w * Vs[d * Dm + h * DHh + k];
  }
  float inv = 1.f / den;
  float* Yo = Y + (size_t)bt * Nn * Dm + i * Dm + h * DHh;
#pragma unroll
  for (int k = 0; k < DHh; k++) Yo[k] = y[k] * inv;
}

// ---------------- node gate (squeeze-excite), in-place on U ----------------
__global__ void k_gate(float* __restrict__ U, const float* __restrict__ g1w,
                       const float* __restrict__ g1b, const float* __restrict__ g2w,
                       const float* __restrict__ g2b_p) {
  int wave = threadIdx.x >> 5, lane = threadIdx.x & 31;
  size_t row = (size_t)blockIdx.x * 8 + wave;
  float4* ur = (float4*)(U + row * Dm);
  float4 v = ur[lane];
  float s = v.x + v.y + v.z + v.w;
#pragma unroll
  for (int o = 16; o; o >>= 1) s += __shfl_xor(s, o, 32);
  float sU = s * (1.f / Dm);
  float4 w1 = ((const float4*)g1w)[lane];
  float4 b1 = ((const float4*)g1b)[lane];
  float4 w2 = ((const float4*)g2w)[lane];
  float t0 = sU * w1.x + b1.x, t1 = sU * w1.y + b1.y;
  float t2 = sU * w1.z + b1.z, t3 = sU * w1.w + b1.w;
  float p = (t0 / (1.f + __expf(-t0))) * w2.x + (t1 / (1.f + __expf(-t1))) * w2.y +
            (t2 / (1.f + __expf(-t2))) * w2.z + (t3 / (1.f + __expf(-t3))) * w2.w;
#pragma unroll
  for (int o = 16; o; o >>= 1) p += __shfl_xor(p, o, 32);
  float g = 1.f / (1.f + __expf(-(p + g2b_p[0])));
  v.x *= g; v.y *= g; v.z *= g; v.w *= g;
  ur[lane] = v;
}

// ---------------- fused MLP: Z = U + silu(Uln@Wm1+b1)@Wm2 + b2 ----------------
// W1T pre-transposed [512][128] (W1T[n][k]); W2T pre-transposed [128][512] (W2T[n][k]).
__global__ void k_mlp(const float* __restrict__ Uln, const bf16* __restrict__ W1T,
                      const float* __restrict__ b1, const bf16* __restrict__ W2T,
                      const float* __restrict__ b2, float* __restrict__ Z) {
  extern __shared__ __align__(16) char smem[];
  bf16* Abf = (bf16*)smem;             // [128][LDK] Uln tile
  bf16* WT  = Abf + 128 * LDK;         // [128][LDK] staged W chunk (transposed)
  bf16* Gbf = WT  + 128 * LDK;         // [128][LDK] hidden tile
  int tid = threadIdx.x;
  size_t rowBase = (size_t)blockIdx.x * 128;
  unsigned wtb = (unsigned)(size_t)WT;

  const float4* Ag = (const float4*)(Uln + rowBase * Dm);
#pragma unroll
  for (int it = 0; it < 16; it++) {
    int f = tid + it * 256;
    int r = f >> 5, c4 = f & 31;
    float4 v = Ag[f];
    bf16* p = Abf + r * LDK + c4 * 4;
    p[0] = (bf16)v.x; p[1] = (bf16)v.y; p[2] = (bf16)v.z; p[3] = (bf16)v.w;
  }
  int wave = tid >> 5, lane = tid & 31, m = lane & 15, lh = lane >> 4;
  v8f acc[8];
#pragma unroll
  for (int ct = 0; ct < 8; ct++) acc[ct] = v8zero();

  for (int c = 0; c < 4; c++) {
    __syncthreads();  // WT free (and Abf staged on first iter)
    // async DMA: W1T chunk rows [c*128 .. c*128+127], contiguous, gstride 128
    stage_wt_async(W1T + (size_t)(c * 128) * 128, 128, wtb, tid);
    wait_async0();
    __syncthreads();
    v8f g8[8];
#pragma unroll
    for (int ct = 0; ct < 8; ct++) g8[ct] = v8zero();
#pragma unroll
    for (int ks = 0; ks < 4; ks++) {
      int k0 = ks * 32;
      v16bf af = load_afrag(Abf + wave * 16 * LDK, m, lh, k0);
#pragma unroll
      for (int ct = 0; ct < 8; ct++) {
        v16bf bfr = load_bfrag(WT, ct * 16 + m, lh, k0);
        g8[ct] = __builtin_amdgcn_wmma_f32_16x16x32_bf16(false, af, false, bfr,
                                                         (short)0, g8[ct], false, false);
      }
    }
    // bias + silu -> Gbf (bf16)
#pragma unroll
    for (int ct = 0; ct < 8; ct++) {
#pragma unroll
      for (int v = 0; v < 8; v++) {
        int rl = wave * 16 + v + 8 * lh;
        int cl = ct * 16 + m;
        float x = g8[ct][v] + b1[c * 128 + cl];
        Gbf[rl * LDK + cl] = (bf16)(x / (1.f + __expf(-x)));
      }
    }
    __syncthreads();  // Gbf ready, all waves done with W1 chunk
    // async DMA: W2T chunk = cols c*128..+127 of each row n, gstride 512
    stage_wt_async(W2T + (size_t)c * 128, 512, wtb, tid);
    wait_async0();
    __syncthreads();
#pragma unroll
    for (int ks = 0; ks < 4; ks++) {
      int k0 = ks * 32;
      v16bf af = load_afrag(Gbf + wave * 16 * LDK, m, lh, k0);
#pragma unroll
      for (int ct = 0; ct < 8; ct++) {
        v16bf bfr = load_bfrag(WT, ct * 16 + m, lh, k0);
        acc[ct] = __builtin_amdgcn_wmma_f32_16x16x32_bf16(false, af, false, bfr,
                                                          (short)0, acc[ct], false, false);
      }
    }
  }
#pragma unroll
  for (int ct = 0; ct < 8; ct++) {
#pragma unroll
    for (int v = 0; v < 8; v++) {
      size_t rg = rowBase + wave * 16 + v + 8 * lh;
      int cg = ct * 16 + m;
      size_t off = rg * Dm + cg;
      Z[off] = Z[off] + acc[ct][v] + b2[cg];
    }
  }
}

// ---------------- output assembly: Zout copy, S = mean over N, A copy ----------------
__global__ void k_out(const float* __restrict__ Z, const float* __restrict__ A,
                      float* __restrict__ out) {
  int b = blockIdx.x, t = threadIdx.x;
  if (b < BTn) {
    const float* Zb = Z + (size_t)b * Nn * Dm;
    float* Ob = out + (size_t)b * Nn * Dm;
    for (int i = t; i < Nn * Dm; i += 128) Ob[i] = Zb[i];
    float s = 0.f;
#pragma unroll
    for (int n = 0; n < Nn; n++) s += Zb[n * Dm + t];
    out[(size_t)Rr * Dm + (size_t)b * Dm + t] = s * (1.f / Nn);
  } else {
    for (int i = t; i < Nn * Nn; i += 128) out[(size_t)Rr * Dm + (size_t)BTn * Dm + i] = A[i];
  }
}

// ---------------- host launcher ----------------
extern "C" void kernel_launch(void* const* d_in, const int* in_sizes, int n_in,
                              void* d_out, int out_size, void* d_ws, size_t ws_size,
                              hipStream_t stream) {
  const float* X    = (const float*)d_in[0];
  const float* Wp   = (const float*)d_in[1];
  const float* bp   = (const float*)d_in[2];
  const float* P    = (const float*)d_in[3];
  const float* Q    = (const float*)d_in[4];
  const float* alp  = (const float*)d_in[5];
  const float* ln1g = (const float*)d_in[6];
  const float* ln1b = (const float*)d_in[7];
  const float* Wlin = (const float*)d_in[8];
  const float* Wval = (const float*)d_in[9];
  const float* aatt = (const float*)d_in[10];
  const float* Wout = (const float*)d_in[11];
  const float* g1w  = (const float*)d_in[12];
  const float* g1b  = (const float*)d_in[13];
  const float* g2w  = (const float*)d_in[14];
  const float* g2b  = (const float*)d_in[15];
  const float* ln2g = (const float*)d_in[16];
  const float* ln2b = (const float*)d_in[17];
  const float* Wm1  = (const float*)d_in[18];
  const float* bm1  = (const float*)d_in[19];
  const float* Wm2  = (const float*)d_in[20];
  const float* bm2  = (const float*)d_in[21];
  const float* A0   = (const float*)d_in[22];
  const int*   src  = (const int*)d_in[24];
  const int*   dst  = (const int*)d_in[25];
  int E = in_sizes[24];

  char* wsp = (char*)d_ws;
  auto alloc = [&](size_t bytes) { char* p = wsp; wsp += (bytes + 255) & ~(size_t)255; return p; };
  float* Zb  = (float*)alloc((size_t)Rr * Dm * 4);
  float* Hn  = (float*)alloc((size_t)Rr * Dm * 4);
  float* Hm  = (float*)alloc((size_t)Rr * Dm * 4);
  float* Xq  = (float*)alloc((size_t)Rr * Dm * 4);
  float* Xv  = (float*)alloc((size_t)Rr * Dm * 4);
  float* Yb  = (float*)alloc((size_t)Rr * Dm * 4);
  float* Aw  = (float*)alloc(Nn * Nn * 4);
  float* lA  = (float*)alloc(Nn * Nn * 4);
  int*   rp  = (int*)alloc((Nn + 1) * 4);
  bf16*  wWpT   = (bf16*)alloc(Dm * Dm * 2);
  bf16*  wWlinT = (bf16*)alloc((size_t)NL * Dm * Dm * 2);
  bf16*  wWvalT = (bf16*)alloc((size_t)NL * Dm * Dm * 2);
  bf16*  wWoutT = (bf16*)alloc((size_t)NL * Dm * Dm * 2);
  bf16*  wWm1T  = (bf16*)alloc((size_t)NL * Dm * HID * 2);
  bf16*  wWm2T  = (bf16*)alloc((size_t)NL * HID * Dm * 2);

  auto cvtT = [&](const float* s, bf16* d, int K, int Ncol) {
    k_f2bfT<<<(K * Ncol + 255) / 256, 256, 0, stream>>>(s, d, K, Ncol);
  };
  cvtT(Wp, wWpT, Dm, Dm);
  for (int l = 0; l < NL; l++) {
    cvtT(Wlin + (size_t)l * Dm * Dm, wWlinT + (size_t)l * Dm * Dm, Dm, Dm);
    cvtT(Wval + (size_t)l * Dm * Dm, wWvalT + (size_t)l * Dm * Dm, Dm, Dm);
    cvtT(Wout + (size_t)l * Dm * Dm, wWoutT + (size_t)l * Dm * Dm, Dm, Dm);
    cvtT(Wm1  + (size_t)l * Dm * HID, wWm1T + (size_t)l * Dm * HID, Dm, HID);   // -> [512][128]
    cvtT(Wm2  + (size_t)l * HID * Dm, wWm2T + (size_t)l * HID * Dm, HID, Dm);   // -> [128][512]
  }

  k_adj<<<1, 256, 0, stream>>>(P, Q, alp, A0, src, dst, E, Aw, lA, rp);

  const size_t gemmLds = (size_t)2 * 128 * LDK * sizeof(bf16);
  const size_t mlpLds  = (size_t)3 * 128 * LDK * sizeof(bf16);
  const size_t attLds  = (size_t)2 * Nn * Dm * sizeof(float);

  // Z = X @ Wp + bp
  k_gemm<<<Rr / 128, 256, gemmLds, stream>>>(X, wWpT, bp, nullptr, nullptr, Zb, 1);

  for (int l = 0; l < NL; l++) {
    k_ln<<<Rr / 8, 256, 0, stream>>>(Zb, ln1g + l * Dm, ln1b + l * Dm, Hn);
    k_hmix<<<BTn, 256, 0, stream>>>(Hn, Aw, Hm);
    k_gemm<<<Rr / 128, 256, gemmLds, stream>>>(Hn, wWlinT + (size_t)l * Dm * Dm,
                                               nullptr, nullptr, nullptr, Xq, 0);
    k_gemm<<<Rr / 128, 256, gemmLds, stream>>>(Hn, wWvalT + (size_t)l * Dm * Dm,
                                               nullptr, nullptr, nullptr, Xv, 0);
    k_attn<<<BTn, 256, attLds, stream>>>(Xq, Xv, aatt + l * NH * DHh, lA, rp, dst, Yb);
    // U = Z + Hmix + Y @ Wout   (written in place over Z)
    k_gemm<<<Rr / 128, 256, gemmLds, stream>>>(Yb, wWoutT + (size_t)l * Dm * Dm,
                                               nullptr, Zb, Hm, Zb, 2 | 4);
    k_gate<<<Rr / 8, 256, 0, stream>>>(Zb, g1w + l * Dm, g1b + l * Dm, g2w + l * Dm, g2b + l);
    k_ln<<<Rr / 8, 256, 0, stream>>>(Zb, ln2g + l * Dm, ln2b + l * Dm, Hn);
    k_mlp<<<Rr / 128, 256, mlpLds, stream>>>(Hn, wWm1T + (size_t)l * Dm * HID, bm1 + l * HID,
                                             wWm2T + (size_t)l * HID * Dm, bm2 + l * Dm, Zb);
  }
  k_out<<<BTn + 1, 128, 0, stream>>>(Zb, Aw, (float*)d_out);
}
